// DimeNet_19146964206349
// MI455X (gfx1250) — matrix operand order repaired and checked by reading, not compiled, not run on hardware
//
#include <hip/hip_runtime.h>
#include <hip/hip_bf16.h>

// ---------------------------------------------------------------------------
// DimeNet interaction block for gfx1250 (MI455X), wave32 + WMMA bf16.
//
//   0. pack all dense weights into per-lane v_wmma bf16 B-fragment layout
//   1. x_ji  = relu(x@W_ji+b)                        (WMMA, bias+relu)
//   2. xkj1  = relu(x@W_kj+b) * rbf_t                (WMMA, rbf_t fused in epilogue)
//   3. xkjI  = relu(xkj1@W_down)                     (WMMA, N=64)
//   4. t8    = sbf@W_sbf1                            (VALU, T x 8)
//   5. agg   = 0 ; scatter: agg[ji] += (t8@W_sbf2)*xkjI[kj]   (atomics, L2-resident)
//   6. h     = relu(agg@W_up) + x_ji                 (WMMA, K=64, add)
//   7..13    residual / lin chain                    (WMMA, bias+relu(+add))
//
// GEMM kernel: one block = one 16-row stripe; the fp32->bf16 A-fragment pack
// is done cooperatively ONCE per block into LDS (not per-wave), so the inner
// loop is ds_load_b128 + global b128 (weights) + v_wmma with minimal VALU.
//
// Workspace need: 512KB (packed W) + 64MB + 32MB + 32MB = 128.5 MB.
// d_out (64MB) doubles as a scratch ping buffer; fully overwritten at the end.
// ---------------------------------------------------------------------------

#define H_    128
#define INT_  64
#define NRAD_ 6
#define SBFD_ 42   // NS*NR

typedef __attribute__((ext_vector_type(16))) __bf16 v16bf;
typedef __attribute__((ext_vector_type(8)))  float  v8f;

union ABfrag { v16bf v; unsigned u[8]; };

__device__ __forceinline__ unsigned pk_bf16(float x, float y) {
  // round-to-nearest-even fp32 -> bf16, packed pair
  unsigned a = __float_as_uint(x);
  unsigned b = __float_as_uint(y);
  a += 0x7FFFu + ((a >> 16) & 1u);
  b += 0x7FFFu + ((b >> 16) & 1u);
  return (a >> 16) | (b & 0xFFFF0000u);
}

// --- pack a row-major fp32 weight [K x N] into WMMA bf16 B fragments --------
// B (32x16 bf16) layout: lane%16 = N column, lane/16 selects K half (16..31),
// dword v holds K = kstep*32 + 2v + 16*(lane/16) + {0,1}.
// Storage: [kstep][ntile][lane][v] dwords -> b128-loadable per lane.
__global__ __launch_bounds__(256) void pack_weight(const float* __restrict__ W,
                                                   unsigned* __restrict__ out,
                                                   int K, int N) {
  int tid = blockIdx.x * 256 + threadIdx.x;
  int total = (K * N) >> 1;
  if (tid >= total) return;
  int v     = tid & 7;
  int lane  = (tid >> 3) & 31;
  int rest  = tid >> 8;
  int ntls  = N >> 4;
  int ntile = rest % ntls;
  int kstep = rest / ntls;
  int col   = ntile * 16 + (lane & 15);
  int half  = lane >> 4;
  int kc    = kstep * 32 + 2 * v + 16 * half;
  out[tid] = pk_bf16(W[kc * N + col], W[(kc + 1) * N + col]);
}

// --- generic WMMA GEMM: out = epi(relu(A@B + bias?)) ------------------------
// One block = one 16-row stripe of A; one wave = one 16x16 output tile.
// A: fp32 [M x K] row-major, packed once per block into LDS bf16 fragments.
// epilogues: *rbf_t (HAS_RBF, fused 6->8->128 basis MLP) and/or +addA.
template<bool HAS_BIAS, bool HAS_ADD, bool HAS_RBF, int K, int N>
__global__ __launch_bounds__(256) void gemm_wmma(
    const float* __restrict__ A, const unsigned* __restrict__ Bp,
    const float* __restrict__ bias, const float* __restrict__ addA,
    const float* __restrict__ rbf, const float* __restrict__ Wr1,
    const float* __restrict__ Wr2, float* __restrict__ out)
{
  constexpr int NK = K / 32;
  __shared__ unsigned lds_af[NK * 256];   // A fragments, 1KB per k-step
  __shared__ float    lds_t8[16][8];
  const int mt   = blockIdx.x;
  const int wave = threadIdx.x >> 5;   // n-tile
  const int lane = threadIdx.x & 31;
  const int c16  = lane & 15;
  const int half = lane >> 4;

  if constexpr (HAS_RBF) {
    // rbf_t8[row][j] = sum_k rbf[row][k] * W_rbf1[k][j], for the 16 block rows
    int i = threadIdx.x;
    if (i < 128) {
      int r = i >> 3, j = i & 7;
      const float* rr = rbf + (size_t)(mt * 16 + r) * NRAD_;
      float s = 0.f;
      #pragma unroll
      for (int k = 0; k < NRAD_; ++k) s += rr[k] * Wr1[k * 8 + j];
      lds_t8[r][j] = s;
    }
  }

  // Cooperative pack: 16 x K fp32 stripe -> bf16 A fragments in LDS, once.
  // A-frag (16x32 bf16): lane%16 = M row, dword v -> K=2v+16*(lane/16)
  // (matches the B-fragment K order).
  const float* Ablk = A + (size_t)mt * 16 * K;
  for (int e = threadIdx.x; e < NK * 256; e += blockDim.x) {
    int ks = e >> 8;
    int l  = (e >> 3) & 31;
    int v  = e & 7;
    int r  = l & 15;
    int h2 = l >> 4;
    int kc = ks * 32 + 2 * v + 16 * h2;
    const float* ap = Ablk + r * K + kc;
    lds_af[e] = pk_bf16(ap[0], ap[1]);
  }
  __syncthreads();

  const unsigned* bbase = Bp + ((size_t)wave * 32 + lane) * 8;
  constexpr int bstride = (N >> 4) * 256;  // dwords per k-step

  v8f acc = {};
  #pragma unroll
  for (int ks = 0; ks < NK; ++ks) {
    ABfrag a, b;
    const uint4* ap4 = (const uint4*)&lds_af[ks * 256 + lane * 8];
    uint4 a0 = ap4[0], a1 = ap4[1];            // 2x ds_load_b128
    a.u[0] = a0.x; a.u[1] = a0.y; a.u[2] = a0.z; a.u[3] = a0.w;
    a.u[4] = a1.x; a.u[5] = a1.y; a.u[6] = a1.z; a.u[7] = a1.w;
    const uint4* bp = (const uint4*)(bbase + ks * bstride);
    uint4 b0 = bp[0], b1 = bp[1];              // 2x global b128 (L0-resident)
    b.u[0] = b0.x; b.u[1] = b0.y; b.u[2] = b0.z; b.u[3] = b0.w;
    b.u[4] = b1.x; b.u[5] = b1.y; b.u[6] = b1.z; b.u[7] = b1.w;
    acc = __builtin_amdgcn_wmma_f32_16x16x32_bf16(false, a.v, false, b.v,
                                                  (short)0, acc, false, false);
  }

  const int ncol = wave * 16 + c16;
  float w2c[8];
  if constexpr (HAS_RBF) {
    #pragma unroll
    for (int j = 0; j < 8; ++j) w2c[j] = Wr2[j * H_ + ncol];
  }
  const float bv = HAS_BIAS ? bias[ncol] : 0.f;
  #pragma unroll
  for (int r = 0; r < 8; ++r) {
    const int mrow = mt * 16 + r + 8 * half;   // C/D: VGPR r -> M=r(+8)
    float v = acc[r] + bv;
    v = v > 0.f ? v : 0.f;
    if constexpr (HAS_RBF) {
      float rt = 0.f;
      #pragma unroll
      for (int j = 0; j < 8; ++j) rt += lds_t8[r + 8 * half][j] * w2c[j];
      v *= rt;
    }
    if constexpr (HAS_ADD) v += addA[(size_t)mrow * N + ncol];
    out[(size_t)mrow * N + ncol] = v;
  }
}

// --- t8 = sbf @ W_sbf1  (T x 42 -> T x 8) -----------------------------------
__global__ __launch_bounds__(256) void sbf_proj(const float* __restrict__ sbf,
                                                const float* __restrict__ W1,
                                                float* __restrict__ t8, int T)
{
  int t = blockIdx.x * 256 + threadIdx.x;
  if (t >= T) return;
  const float* s = sbf + (size_t)t * SBFD_;
  float acc[8] = {};
  #pragma unroll 6
  for (int k = 0; k < SBFD_; ++k) {
    float sv = s[k];
    #pragma unroll
    for (int j = 0; j < 8; ++j) acc[j] += sv * W1[k * 8 + j];
  }
  float* o = t8 + (size_t)t * 8;
  #pragma unroll
  for (int j = 0; j < 8; ++j) o[j] = acc[j];
}

// --- scatter: agg[ji] += (t8 @ W_sbf2) * xkjI[kj]  (4 threads / triplet) ----
__global__ __launch_bounds__(256) void triplet_scatter(
    const float* __restrict__ t8, const float* __restrict__ W2,
    const float* __restrict__ xk, const int* __restrict__ idx_kj,
    const int* __restrict__ idx_ji, float* __restrict__ agg, int T)
{
  int gid = blockIdx.x * 256 + threadIdx.x;
  int t = gid >> 2, q = gid & 3;
  if (t >= T) return;
  float a[8];
  const float* tp = t8 + (size_t)t * 8;
  #pragma unroll
  for (int j = 0; j < 8; ++j) a[j] = tp[j];
  const int kj = idx_kj[t], ji = idx_ji[t];
  const float* g = xk  + (size_t)kj * INT_ + q * 16;
  float*       o = agg + (size_t)ji * INT_ + q * 16;
  #pragma unroll
  for (int c = 0; c < 16; ++c) {
    float s = 0.f;
    #pragma unroll
    for (int j = 0; j < 8; ++j) s += a[j] * W2[j * INT_ + q * 16 + c];
    atomicAdd(o + c, s * g[c]);     // global_atomic_add_f32, agg is L2-resident
  }
}

extern "C" void kernel_launch(void* const* d_in, const int* in_sizes, int n_in,
                              void* d_out, int out_size, void* d_ws, size_t ws_size,
                              hipStream_t stream)
{
  const float* x      = (const float*)d_in[0];
  const float* rbf    = (const float*)d_in[1];
  const float* sbf    = (const float*)d_in[2];
  const int*   idx_kj = (const int*)  d_in[3];
  const int*   idx_ji = (const int*)  d_in[4];
  const float* W_rbf1 = (const float*)d_in[5];
  const float* W_rbf2 = (const float*)d_in[6];
  const float* W_sbf1 = (const float*)d_in[7];
  const float* W_sbf2 = (const float*)d_in[8];
  const float* W_kj   = (const float*)d_in[9];
  const float* b_kj   = (const float*)d_in[10];
  const float* W_ji   = (const float*)d_in[11];
  const float* b_ji   = (const float*)d_in[12];
  const float* W_down = (const float*)d_in[13];
  const float* W_up   = (const float*)d_in[14];
  const float* W_bs   = (const float*)d_in[15];  // (1,2,128,128)
  const float* b_bs   = (const float*)d_in[16];  // (1,2,128)
  const float* W_lin  = (const float*)d_in[17];
  const float* b_lin  = (const float*)d_in[18];
  const float* W_as   = (const float*)d_in[19];  // (2,2,128,128)
  const float* b_as   = (const float*)d_in[20];  // (2,2,128)

  const int E  = in_sizes[0] / H_;      // 131072
  const int T  = in_sizes[3];           // 1048576
  const int MT = E >> 4;                // m-tiles

  // ---- workspace carve-up (needs ~128.5 MB) ----
  unsigned* WP = (unsigned*)d_ws;                 // packed weights, <512 KB
  char* base = (char*)d_ws + 512 * 1024;
  float* P0 = (float*)base;                                   // 64 MB ping
  float* C  = (float*)(base + (size_t)E * H_ * 4);            // 32 MB xkjI
  float* D  = (float*)(base + (size_t)E * H_ * 4
                            + (size_t)E * INT_ * 4);          // 32 MB agg
  float* U3 = C;                     // u3 reuses contiguous C+D (64 MB)
  float* OUT = (float*)d_out;        // 64 MB, doubles as ping buffer

  // packed weight offsets (dwords)
  const int wji = 0,     wkj = 8192,  wdn = 16384, wup = 20480;
  const int wb0 = 24576, wb1 = 32768, wln = 40960;
  const int wa00 = 49152, wa01 = 57344, wa10 = 65536, wa11 = 73728;

  auto packW = [&](const float* W, int off, int K, int N) {
    int total = (K * N) >> 1;
    pack_weight<<<(total + 255) / 256, 256, 0, stream>>>(W, WP + off, K, N);
  };
  packW(W_ji,          wji,  128, 128);
  packW(W_kj,          wkj,  128, 128);
  packW(W_down,        wdn,  128,  64);
  packW(W_up,          wup,   64, 128);
  packW(W_bs + 0,      wb0,  128, 128);
  packW(W_bs + 16384,  wb1,  128, 128);
  packW(W_lin,         wln,  128, 128);
  packW(W_as + 0,      wa00, 128, 128);
  packW(W_as + 16384,  wa01, 128, 128);
  packW(W_as + 32768,  wa10, 128, 128);
  packW(W_as + 49152,  wa11, 128, 128);

  // 1. x_ji = relu(x@W_ji + b)                  -> P0
  gemm_wmma<true, false, false, 128, 128><<<MT, 256, 0, stream>>>(
      x, WP + wji, b_ji, nullptr, nullptr, nullptr, nullptr, P0);
  // 2. xkj1 = relu(x@W_kj + b) * rbf_t          -> OUT (d_out as scratch)
  gemm_wmma<true, false, true, 128, 128><<<MT, 256, 0, stream>>>(
      x, WP + wkj, b_kj, nullptr, rbf, W_rbf1, W_rbf2, OUT);
  // 3. xkjI = relu(xkj1@W_down)                 -> C   (N=64, 4 waves)
  gemm_wmma<false, false, false, 128, 64><<<MT, 128, 0, stream>>>(
      OUT, WP + wdn, nullptr, nullptr, nullptr, nullptr, nullptr, C);
  // 4. t8 = sbf@W_sbf1                          -> OUT (xkj1 dead)
  sbf_proj<<<T / 256, 256, 0, stream>>>(sbf, W_sbf1, OUT, T);
  // 5. agg = 0 ; scatter-add                    -> D
  hipMemsetAsync(D, 0, (size_t)E * INT_ * 4, stream);
  triplet_scatter<<<(4 * T) / 256, 256, 0, stream>>>(
      OUT, W_sbf2, C, idx_kj, idx_ji, D, T);
  // 6. h = relu(agg@W_up) + x_ji                -> OUT (t8 dead), K=64
  gemm_wmma<false, true, false, 64, 128><<<MT, 256, 0, stream>>>(
      D, WP + wup, nullptr, P0, nullptr, nullptr, nullptr, OUT);
  // 7. u = relu(h@Wbs0 + b)                     -> P0 (x_ji dead)
  gemm_wmma<true, false, false, 128, 128><<<MT, 256, 0, stream>>>(
      OUT, WP + wb0, b_bs + 0, nullptr, nullptr, nullptr, nullptr, P0);
  // 8. h1 = h + relu(u@Wbs1 + b)                -> OUT (in-place add, A=P0)
  gemm_wmma<true, true, false, 128, 128><<<MT, 256, 0, stream>>>(
      P0, WP + wb1, b_bs + 128, OUT, nullptr, nullptr, nullptr, OUT);
  // 9. h2 = relu(h1@W_lin + b) + x              -> P0
  gemm_wmma<true, true, false, 128, 128><<<MT, 256, 0, stream>>>(
      OUT, WP + wln, b_lin, x, nullptr, nullptr, nullptr, P0);
  // 10. u2 = relu(h2@Was00 + b)                 -> OUT
  gemm_wmma<true, false, false, 128, 128><<<MT, 256, 0, stream>>>(
      P0, WP + wa00, b_as + 0, nullptr, nullptr, nullptr, nullptr, OUT);
  // 11. h3 = h2 + relu(u2@Was01 + b)            -> P0 (in-place add, A=OUT)
  gemm_wmma<true, true, false, 128, 128><<<MT, 256, 0, stream>>>(
      OUT, WP + wa01, b_as + 128, P0, nullptr, nullptr, nullptr, P0);
  // 12. u3 = relu(h3@Was10 + b)                 -> U3 (= C+D, both dead)
  gemm_wmma<true, false, false, 128, 128><<<MT, 256, 0, stream>>>(
      P0, WP + wa10, b_as + 256, nullptr, nullptr, nullptr, nullptr, U3);
  // 13. out = h3 + relu(u3@Was11 + b)           -> d_out (final)
  gemm_wmma<true, true, false, 128, 128><<<MT, 256, 0, stream>>>(
      U3, WP + wa11, b_as + 384, P0, nullptr, nullptr, nullptr, OUT);
}
